// MiniRocket_29188597744079
// MI455X (gfx1250) — matrix operand order repaired and checked by reading, not compiled, not run on hardware
//
#include <hip/hip_runtime.h>
#include <cmath>
#include <cstdint>

// ---------------------------------------------------------------------------
// MiniRocket transform for MI455X (gfx1250, wave32).
//  * C-matrix formation (9-tap dilated conv, 84 kernels) -> v_wmma_f32_16x16x32_f16
//  * series staging global->LDS -> Tensor Data Mover (tensor_load_to_lds)
//  * PPV threshold-counting -> VALU, fully unrolled via template<int F>
// ---------------------------------------------------------------------------

typedef __attribute__((ext_vector_type(16))) _Float16 v16h;
typedef __attribute__((ext_vector_type(8)))  float    v8f;
typedef __attribute__((ext_vector_type(4)))  uint32_t u32x4;
typedef __attribute__((ext_vector_type(8)))  int      i32x8;
typedef __attribute__((ext_vector_type(4)))  int      i32x4;

#define KLEN   9
#define NKER   84
#define SLEN   1024
#define MAXD   32
#define MAXF   20

#if defined(__HIP_DEVICE_COMPILE__) && defined(__has_builtin)
#if __has_builtin(__builtin_amdgcn_tensor_load_to_lds) && \
    __has_builtin(__builtin_amdgcn_s_wait_tensorcnt)
#define USE_TDM 1
#endif
#endif

struct GroupPlan {          // dilations sharing the same features-per-kernel F
    int n;
    int dil[MAXD];          // dilation value
    int off[MAXD];          // offset into biases / feature axis
    int par[MAXD];          // di % 2 (parity for padded/unpadded selection)
};

template <int F>
__global__ __launch_bounds__(192)
void minirocket_wmma_kernel(const float* __restrict__ x,
                            const float* __restrict__ biases,
                            float* __restrict__ out,
                            int nfeat, GroupPlan gp)
{
    // f32 copy of one series, zero-padded by p=4d each side (max 2040 floats).
    __shared__ float xs[2048];

    const int tid    = threadIdx.x;
    const int lane   = tid & 31;
    const int kt     = tid >> 5;            // kernel tile 0..5 (one per wave)
    const int gi     = blockIdx.x;          // dilation within this F-group
    const int series = blockIdx.y;

    const int d   = gp.dil[gi];
    const int off = gp.off[gi];
    const int par = gp.par[gi];
    const int p   = 4 * d;                  // 'same' padding each side

    __builtin_prefetch(biases + off, 0, 3); // global_prefetch_b8

    // ---- Stage series into LDS (zeros cover the pad regions) ---------------
    for (int i = tid; i < 2048; i += 192) xs[i] = 0.0f;
    __syncthreads();
    const float* xsrc = x + (size_t)series * SLEN;
#ifdef USE_TDM
    if (kt == 0) {                          // one wave drives the TDM DMA
        const uint32_t lds_byte = (uint32_t)(size_t)(&xs[p]);
        const uint64_t ga       = (uint64_t)(size_t)xsrc;
        u32x4 g0;
        g0.x = 1u;                                          // count=1 (valid D#)
        g0.y = lds_byte;                                    // lds_addr (bytes)
        g0.z = (uint32_t)(ga & 0xffffffffu);                // global_addr[31:0]
        g0.w = (uint32_t)((ga >> 32) & 0x01ffffffu)         // global_addr[56:32]
             | 0x80000000u;                                 // type=2 ("image")
        // data_size=4B; tensor 1024x1, tile 1024x1, dim0_stride=1024.
        i32x8 g1 = { 0x00020000, 0x04000000, 0x00010000, 0x04000000,
                     1, 1024, 0, 0 };
        i32x4 z4 = { 0, 0, 0, 0 };
        i32x8 z8 = { 0, 0, 0, 0, 0, 0, 0, 0 };
        __builtin_amdgcn_tensor_load_to_lds(g0, g1, z4, z4, z8, 0);
        __builtin_amdgcn_s_wait_tensorcnt(0);
    }
    __syncthreads();
#else
    for (int i = tid; i < SLEN; i += 192) xs[p + i] = xsrc[i];
    __syncthreads();
#endif

    // ---- Per-lane roles -----------------------------------------------------
    const int  n      = lane & 15;          // A row (time) / B & D column (kernel)
    const int  hi     = lane >> 4;          // half-wave selector
    const int  kg     = kt * 16 + n;        // global kernel id
    const bool kvalid = (kg < NKER);

    // ---- B operand: taps 0..4 at K=0..4, taps 5..8 at K=9..12, K=8 dead -----
    // (K=8 receives a duplicated tap-4 from the high half of A; its weight row
    //  is zero so it contributes nothing.)
    int ca = 0, cb = 1, cc = 2;
    if (kvalid) {
        int idx = 0; bool done = false;
        for (int a = 0; a < 7 && !done; ++a)
            for (int b = a + 1; b < 8 && !done; ++b)
                for (int c = b + 1; c < 9; ++c) {
                    if (idx == kg) { ca = a; cb = b; cc = c; done = true; break; }
                    ++idx;
                }
    }
    v16h B = {};
    if (kvalid) {                           // lo lanes: taps 0..4 at s=0..4; hi lanes: taps 5..8 at s=1..4
        for (int e = hi * 5; e < 5 + hi * 4; ++e) {
            float w = (e == ca || e == cb || e == cc) ? 2.0f : -1.0f;
            B[e - 4 * hi] = (_Float16)w;
        }
    }

    // ---- PPV range + bias preload (all in registers: F is compile-time) -----
    const bool full  = kvalid && (((par + kg) & 1) == 0);
    const int  lo    = full ? 0 : p;
    const unsigned range = full ? SLEN : (SLEN - 2 * p);
    const float norm = (float)range;

    float bias[F];
#pragma unroll
    for (int j = 0; j < F; ++j)
        bias[j] = kvalid ? biases[off + kg * F + j] : 3.0e38f;

    float cnt[F];
#pragma unroll
    for (int j = 0; j < F; ++j) cnt[j] = 0.0f;

    // ---- Time loop: one WMMA per 16x16 (time x kernel) tile -----------------
    const int ebase = hi * 4;               // lo lanes: taps 0..4, hi: taps 4..8
    for (int t0 = 0; t0 < SLEN; t0 += 16) {
        const int base = t0 + n;            // xp index of tap j is base + j*d
        v16h A = {};
#pragma unroll
        for (int e = 0; e < 5; ++e)         // uniform: 5 LDS loads per lane
            A[e] = (_Float16)xs[base + (e + ebase) * d];

        v8f c = {};
        c = __builtin_amdgcn_wmma_f32_16x16x32_f16(
                false, A, false, B, (short)0, c, false, false);

        // D layout: lane owns kernel col n, time rows t0 + v + 8*hi.
        float dv[8];
        const bool interior = (t0 >= p) && (t0 + 16 <= SLEN - p);
        if (interior) {
#pragma unroll
            for (int v = 0; v < 8; ++v) dv[v] = c[v];
        } else {
            const int tb = t0 + 8 * hi - lo;
#pragma unroll
            for (int v = 0; v < 8; ++v)
                dv[v] = ((unsigned)(tb + v) < range) ? c[v] : -3.0e38f;
        }

#pragma unroll
        for (int j = 0; j < F; ++j) {
            float s = 0.0f;
#pragma unroll
            for (int v = 0; v < 8; ++v)
                s += (dv[v] > bias[j]) ? 1.0f : 0.0f;
            cnt[j] += s;
        }
    }

    // ---- Merge the two row-halves, normalize, store -------------------------
#pragma unroll
    for (int j = 0; j < F; ++j) {
        const float tot = cnt[j] + __shfl_xor(cnt[j], 16, 32);
        if (hi == 0 && kvalid)
            out[(size_t)series * nfeat + off + kg * F + j] = tot / norm;
    }
}

// ---------------------------------------------------------------------------
// Host: deterministic replication of MiniRocket _fit_dilations (numpy doubles)
// ---------------------------------------------------------------------------
struct DilPlan { int nd; int dil[MAXD]; int nf[MAXD]; int off[MAXD]; };

static void make_plan(DilPlan& P)
{
    const int    nfpk       = 9996 / NKER;          // 119
    const int    true_max   = 32;                   // min(119, 32)
    const double multiplier = (double)nfpk / (double)true_max;
    const double max_exp    = std::log2((double)(SLEN - 1) / (double)(KLEN - 1));

    int vals[32];
    for (int k = 0; k < 32; ++k) {
        const double e = max_exp * (double)k / 31.0;
        vals[k] = (int)std::pow(2.0, e);            // truncation = astype(int32)
    }
    int nd = 0, dil[32], cnts[32];
    for (int k = 0; k < 32; ++k) {
        if (nd > 0 && dil[nd - 1] == vals[k]) { cnts[nd - 1]++; }
        else { dil[nd] = vals[k]; cnts[nd] = 1; ++nd; }
    }
    int nf[32], sum = 0;
    for (int i = 0; i < nd; ++i) { nf[i] = (int)((double)cnts[i] * multiplier); sum += nf[i]; }
    int rem = nfpk - sum, i = 0;
    while (rem > 0) { nf[i] += 1; --rem; i = (i + 1) % nd; }

    int off = 0;
    for (int j = 0; j < nd; ++j) {
        P.dil[j] = dil[j];
        P.nf[j]  = nf[j] > MAXF ? MAXF : nf[j];     // defensive clamp (19 expected)
        P.off[j] = off;
        off += NKER * nf[j];
    }
    P.nd = nd;
}

extern "C" void kernel_launch(void* const* d_in, const int* in_sizes, int n_in,
                              void* d_out, int out_size, void* d_ws, size_t ws_size,
                              hipStream_t stream)
{
    (void)n_in; (void)d_ws; (void)ws_size;
    const float* x      = (const float*)d_in[0];   // (N, 1, 1024) f32
    const float* biases = (const float*)d_in[1];   // (9996,) f32
    float*       out    = (float*)d_out;           // (N, 9996) f32

    const int N     = in_sizes[0] / SLEN;          // 256
    const int nfeat = out_size / N;                // 9996

    DilPlan plan;
    make_plan(plan);

    // Group dilations by features-per-kernel F; one templated launch per group.
    bool used[MAXD] = {};
    for (int i = 0; i < plan.nd; ++i) {
        if (used[i]) continue;
        const int F = plan.nf[i];
        GroupPlan gp; gp.n = 0;
        for (int j = i; j < plan.nd; ++j) {
            if (!used[j] && plan.nf[j] == F) {
                gp.dil[gp.n] = plan.dil[j];
                gp.off[gp.n] = plan.off[j];
                gp.par[gp.n] = j & 1;
                gp.n++;
                used[j] = true;
            }
        }
        dim3 grid((unsigned)gp.n, (unsigned)N, 1);
        dim3 block(192, 1, 1);                     // 6 waves = 6 kernel tiles
#define LAUNCH_F(NF) case NF: \
        minirocket_wmma_kernel<NF><<<grid, block, 0, stream>>>(x, biases, out, nfeat, gp); break;
        switch (F) {
            LAUNCH_F(1)  LAUNCH_F(2)  LAUNCH_F(3)  LAUNCH_F(4)  LAUNCH_F(5)
            LAUNCH_F(6)  LAUNCH_F(7)  LAUNCH_F(8)  LAUNCH_F(9)  LAUNCH_F(10)
            LAUNCH_F(11) LAUNCH_F(12) LAUNCH_F(13) LAUNCH_F(14) LAUNCH_F(15)
            LAUNCH_F(16) LAUNCH_F(17) LAUNCH_F(18) LAUNCH_F(19) LAUNCH_F(20)
            default: break;                        // cannot occur (clamped)
        }
#undef LAUNCH_F
    }
}
